// SetAbstraction_63170378990116
// MI455X (gfx1250) — compile-verified, matrix operand order checked by reading, not run on hardware
//
#include <hip/hip_runtime.h>
#include <hip/hip_bf16.h>
#include <cstdint>

#define NBALLS 2048
#define KNN 65          // K+1 neighbors kept
#define RADIUSF 0.2f

typedef __attribute__((ext_vector_type(16))) _Float16 v16h;
typedef __attribute__((ext_vector_type(8)))  float    v8f;
typedef __attribute__((ext_vector_type(2)))  float    v2f;

__device__ __forceinline__ uint32_t lds_addr32(const void* p) {
    // LDS aperture: the low 32 bits of the pointer give the LDS address
    return (uint32_t)(uintptr_t)p;
}

__device__ __forceinline__ void async_load_b128_to_lds(uint32_t lds_off, const void* gaddr) {
    asm volatile("global_load_async_to_lds_b128 %0, %1, off"
                 :: "v"(lds_off), "v"(gaddr) : "memory");
}

__device__ __forceinline__ void wait_asynccnt0() {
    asm volatile("s_wait_asynccnt 0x0" ::: "memory");
}

// ---------------------------------------------------------------------------
// Kernel 1: farthest point sampling. One persistent workgroup; mind2 lives in
// dynamic LDS (50000 floats = 200 KB <= 320 KB WGP LDS) so the 2047 serial
// argmax+update sweeps never leave the WGP.
// ---------------------------------------------------------------------------
__global__ void __launch_bounds__(1024)
fps_kernel(const float* __restrict__ pts, int V, int* __restrict__ cidx)
{
    extern __shared__ float mind2[];     // V floats (dynamic LDS)
    __shared__ float sval[1024];
    __shared__ int   sidx[1024];
    const int tid = threadIdx.x;

    const float p0x = pts[0], p0y = pts[1], p0z = pts[2];
    for (int j = tid; j < V; j += 1024) {
        const float dx = pts[3*j] - p0x, dy = pts[3*j+1] - p0y, dz = pts[3*j+2] - p0z;
        mind2[j] = dx*dx + dy*dy + dz*dz;
    }
    if (tid == 0) cidx[0] = 0;
    __syncthreads();

    for (int s = 1; s < NBALLS; ++s) {
        // argmax over mind2 (ties -> lowest index, matching jnp.argmax)
        float bv = -1.0f; int bi = 0x7fffffff;
        for (int j = tid; j < V; j += 1024) {
            const float v = mind2[j];
            if (v > bv || (v == bv && j < bi)) { bv = v; bi = j; }
        }
        sval[tid] = bv; sidx[tid] = bi;
        __syncthreads();
        for (int off = 512; off > 0; off >>= 1) {
            if (tid < off) {
                const float ov = sval[tid+off]; const int oi = sidx[tid+off];
                if (ov > sval[tid] || (ov == sval[tid] && oi < sidx[tid])) {
                    sval[tid] = ov; sidx[tid] = oi;
                }
            }
            __syncthreads();
        }
        const int idx = sidx[0];
        if (tid == 0) cidx[s] = idx;
        const float qx = pts[3*idx], qy = pts[3*idx+1], qz = pts[3*idx+2];
        for (int j = tid; j < V; j += 1024) {
            const float dx = pts[3*j]-qx, dy = pts[3*j+1]-qy, dz = pts[3*j+2]-qz;
            mind2[j] = fminf(mind2[j], dx*dx + dy*dy + dz*dz);
        }
        __syncthreads();
    }
}

// ---------------------------------------------------------------------------
// Kernel 2: 65-NN, one wave per centroid. Each lane scans a 32-strided slice
// (consecutive lanes -> consecutive points -> coalesced), keeps a private
// sorted top-65, then a 32-way merge via shfl_xor argmin (tie -> lowest point
// index) reproduces jax.lax.top_k's stable ordering exactly.
// Distance is the reference's exact formula: d = sqrt(|2*sn_j - 2*c.x_j|).
// ---------------------------------------------------------------------------
__global__ void __launch_bounds__(64)
knn_kernel(const float* __restrict__ pts, int V, const int* __restrict__ cidx,
           int* __restrict__ nidx, float* __restrict__ ndist)
{
    __shared__ float cd[2][32][KNN];
    __shared__ int   cj[2][32][KNN];

    const int wave = threadIdx.x >> 5;
    const int lane = threadIdx.x & 31;
    const int c    = blockIdx.x * 2 + wave;

    const int ci = cidx[c];
    const float cx = pts[3*ci], cy = pts[3*ci+1], cz = pts[3*ci+2];

    float best[KNN]; int besti[KNN];
    for (int k = 0; k < KNN; ++k) { best[k] = 3.0e38f; besti[k] = 0; }

    for (int j = lane; j < V; j += 32) {
        if (j + 256 < V) __builtin_prefetch(&pts[3*(j + 256)], 0, 1);
        const float px = pts[3*j], py = pts[3*j+1], pz = pts[3*j+2];
        const float sn  = px*px + py*py + pz*pz;
        const float dot = cx*px + cy*py + cz*pz;
        const float d = sqrtf(fabsf(2.0f*sn - 2.0f*dot));
        if (d < best[KNN-1]) {
            int p = KNN-1;
            while (p > 0 && best[p-1] > d) {  // strict: equal d keeps earlier j first
                best[p] = best[p-1]; besti[p] = besti[p-1]; --p;
            }
            best[p] = d; besti[p] = j;
        }
    }
    for (int k = 0; k < KNN; ++k) { cd[wave][lane][k] = best[k]; cj[wave][lane][k] = besti[k]; }
    __syncthreads();

    // 32-way merge of sorted per-lane lists
    int p = 0;
    for (int k = 0; k < KNN; ++k) {
        float hd = cd[wave][lane][p];
        int   hj = cj[wave][lane][p];
        float bd = hd; int bj = hj; int bl = lane;
        for (int off = 16; off > 0; off >>= 1) {
            const float od = __shfl_xor(bd, off);
            const int   oj = __shfl_xor(bj, off);
            const int   ol = __shfl_xor(bl, off);
            if (od < bd || (od == bd && oj < bj)) { bd = od; bj = oj; bl = ol; }
        }
        if (lane == 0) { ndist[c*KNN + k] = bd; nidx[c*KNN + k] = bj; }
        if (lane == bl) ++p;
    }
}

// ---------------------------------------------------------------------------
// Kernel 3: per-centroid MLP (6->64 relu, 64->64) via v_wmma_f32_16x16x32_f16
//           + masked max aggregation -> agg[2048,64]
// W1/W2 are staged into LDS once per block with gfx1250 async-to-LDS copies.
// Block = 5 waves (160 thr); wave w owns neighbor-row tile [16w, 16w+16)
// ---------------------------------------------------------------------------
__global__ void __launch_bounds__(160)
mlp_kernel(const float* __restrict__ pts, const int* __restrict__ cidx,
           const int* __restrict__ nidx, const float* __restrict__ ndist,
           const float* __restrict__ W1, const float* __restrict__ b1,
           const float* __restrict__ W2, const float* __restrict__ b2,
           float* __restrict__ aggout)
{
    __shared__ __align__(16) float w1s[6*64];    // 1.5 KB
    __shared__ __align__(16) float w2s[64*64];   // 16 KB
    __shared__ _Float16 hbuf[80][64];            // relu(layer1) rows (f16)
    __shared__ float    wmax[5][64];             // per-wave partial column maxima
    __shared__ unsigned char svalid[80];         // neighbor validity mask

    const int c    = blockIdx.x;
    const int tid  = threadIdx.x;
    const int wave = tid >> 5;
    const int lane = tid & 31;
    const int l16  = lane & 15;
    const bool lo  = lane < 16;

    // ---- async-stage weights into LDS (ASYNCcnt-tracked) ----
    {
        const uint32_t w1base = lds_addr32(&w1s[0]);
        for (int i = tid; i < (6*64)/4; i += 160)
            async_load_b128_to_lds(w1base + i*16, W1 + i*4);
        const uint32_t w2base = lds_addr32(&w2s[0]);
        for (int i = tid; i < (64*64)/4; i += 160)
            async_load_b128_to_lds(w2base + i*16, W2 + i*4);
        wait_asynccnt0();
    }

    if (tid < 80) {
        const float limit = fminf(ndist[c*KNN + (KNN-1)], RADIUSF);
        svalid[tid] = (tid < KNN) ? ((ndist[c*KNN + tid] <= limit) ? 1 : 0) : 0;
    }

    const int ci = cidx[c];
    const float ccx = pts[3*ci], ccy = pts[3*ci+1], ccz = pts[3*ci+2];

    // A1 fragment: 16x32 f16. Lanes 0-15 hold K={0..7,16..23} of row M=lane;
    // lanes 16-31 hold K={8..15,24..31} which are all zero padding (K only 0..5).
    v16h a1; for (int i = 0; i < 16; ++i) a1[i] = (_Float16)0.0f;
    if (lo) {
        const int row = wave*16 + lane;
        if (row < KNN) {
            const int j = nidx[c*KNN + row];
            const float px = pts[3*j], py = pts[3*j+1], pz = pts[3*j+2];
            a1[0] = (_Float16)px;        a1[1] = (_Float16)py;        a1[2] = (_Float16)pz;
            a1[3] = (_Float16)(px-ccx);  a1[4] = (_Float16)(py-ccy);  a1[5] = (_Float16)(pz-ccz);
        }
    }
    __syncthreads();   // weights staged + svalid ready

    // ---- layer 1: h = relu(feat @ W1 + b1), 4 column tiles of 16 ----
    for (int t = 0; t < 4; ++t) {
        v16h bf; for (int i = 0; i < 16; ++i) bf[i] = (_Float16)0.0f;
        if (lo) {
            const int n = t*16 + lane;
            for (int k = 0; k < 6; ++k) bf[k] = (_Float16)w1s[k*64 + n];
        }
        const float bias = b1[t*16 + l16];
        v8f acc; for (int i = 0; i < 8; ++i) acc[i] = bias;
        acc = __builtin_amdgcn_wmma_f32_16x16x32_f16(false, a1, false, bf,
                                                     (short)0, acc, false, false);
        const int mofs = lo ? 0 : 8;   // C/D layout: VGPR r -> M=r (lo) / M=8+r (hi)
        for (int r = 0; r < 8; ++r)
            hbuf[wave*16 + mofs + r][t*16 + l16] = (_Float16)fmaxf(acc[r], 0.0f);
    }
    __syncthreads();

    // ---- layer 2: m = h @ W2 + b2, K=64 as two 16x16x32 steps ----
    v16h a2c[2];
    const int arow = wave*16 + l16;
    for (int kc = 0; kc < 2; ++kc) {
        const int k0 = kc*32 + (lo ? 0 : 8);
        for (int i = 0; i < 8; ++i) {
            a2c[kc][i]     = hbuf[arow][k0 + i];
            a2c[kc][8 + i] = hbuf[arow][k0 + 16 + i];
        }
    }
    for (int t = 0; t < 4; ++t) {
        const int n = t*16 + l16;
        const float bias2 = b2[n];
        v8f acc; for (int i = 0; i < 8; ++i) acc[i] = bias2;
        for (int kc = 0; kc < 2; ++kc) {
            v16h bf;
            const int k0 = kc*32 + (lo ? 0 : 8);
            for (int i = 0; i < 8; ++i) {
                bf[i]     = (_Float16)w2s[(k0 + i)      *64 + n];
                bf[8 + i] = (_Float16)w2s[(k0 + 16 + i) *64 + n];
            }
            acc = __builtin_amdgcn_wmma_f32_16x16x32_f16(false, a2c[kc], false, bf,
                                                         (short)0, acc, false, false);
        }
        // masked max over this wave's 16 rows for column n
        const int mofs = lo ? 0 : 8;
        float pm = -INFINITY;
        for (int r = 0; r < 8; ++r) {
            const int row = wave*16 + mofs + r;
            if (svalid[row]) pm = fmaxf(pm, acc[r]);
        }
        pm = fmaxf(pm, __shfl_xor(pm, 16));  // merge M{0..7} with M{8..15}
        if (lo) wmax[wave][n] = pm;
    }
    __syncthreads();

    if (tid < 64) {
        float m = wmax[0][tid];
        for (int w = 1; w < 5; ++w) m = fmaxf(m, wmax[w][tid]);
        aggout[c*64 + tid] = m;
    }
}

// ---------------------------------------------------------------------------
// Kernel 4: out = agg @ Wg + bg in full fp32 via v_wmma_f32_16x16x4_f32
// Grid: 128 blocks (16 centroids each) x 8 waves (one 16-col tile each)
// ---------------------------------------------------------------------------
__global__ void __launch_bounds__(256)
final_gemm(const float* __restrict__ agg, const float* __restrict__ Wg,
           const float* __restrict__ bg, float* __restrict__ out)
{
    const int tid  = threadIdx.x;
    const int wave = tid >> 5;
    const int lane = tid & 31;
    const int l16  = lane & 15;
    const bool lo  = lane < 16;
    const int rowbase = blockIdx.x * 16;
    const int n = wave*16 + l16;

    const float bias = bg[n];
    v8f acc; for (int i = 0; i < 8; ++i) acc[i] = bias;

    const int arow = rowbase + l16;
    for (int kc = 0; kc < 16; ++kc) {
        // 32-bit A 16x4 layout: lanes 0-15 hold K={0,1}, lanes 16-31 hold K={2,3}
        const int k0 = kc*4 + (lo ? 0 : 2);
        v2f a; a[0] = agg[arow*64 + k0];     a[1] = agg[arow*64 + k0 + 1];
        v2f b; b[0] = Wg[k0*128 + n];        b[1] = Wg[(k0 + 1)*128 + n];
        acc = __builtin_amdgcn_wmma_f32_16x16x4_f32(false, a, false, b,
                                                    (short)0, acc, false, false);
    }
    const int mofs = lo ? 0 : 8;
    for (int r = 0; r < 8; ++r)
        out[(rowbase + mofs + r)*128 + n] = acc[r];
}

// ---------------------------------------------------------------------------
extern "C" void kernel_launch(void* const* d_in, const int* in_sizes, int n_in,
                              void* d_out, int out_size, void* d_ws, size_t ws_size,
                              hipStream_t stream)
{
    (void)n_in; (void)out_size; (void)ws_size;
    const float* verts = (const float*)d_in[0];
    const float* W1    = (const float*)d_in[1];
    const float* b1    = (const float*)d_in[2];
    const float* W2    = (const float*)d_in[3];
    const float* b2    = (const float*)d_in[4];
    const float* Wg    = (const float*)d_in[5];
    const float* bg    = (const float*)d_in[6];
    const int V = in_sizes[0] / 3;

    char* ws = (char*)d_ws;
    size_t off = 0;
    auto alloc = [&](size_t bytes) -> void* {
        void* p = ws + off;
        off = (off + bytes + 255) & ~size_t(255);
        return p;
    };
    int*   cidx  = (int*)  alloc(sizeof(int)   * NBALLS);
    int*   nidx  = (int*)  alloc(sizeof(int)   * NBALLS * KNN);
    float* ndist = (float*)alloc(sizeof(float) * NBALLS * KNN);
    float* aggb  = (float*)alloc(sizeof(float) * NBALLS * 64);

    const size_t fps_lds = sizeof(float) * (size_t)V;   // mind2 in dynamic LDS (200 KB)

    fps_kernel <<<1,          1024, fps_lds, stream>>>(verts, V, cidx);
    knn_kernel <<<NBALLS/2,   64,   0,       stream>>>(verts, V, cidx, nidx, ndist);
    mlp_kernel <<<NBALLS,     160,  0,       stream>>>(verts, cidx, nidx, ndist,
                                                       W1, b1, W2, b2, aggb);
    final_gemm <<<NBALLS/16,  256,  0,       stream>>>(aggb, Wg, bg, (float*)d_out);
}
